// SoftAttention_9706626089641
// MI455X (gfx1250) — matrix operand order repaired and checked by reading, not compiled, not run on hardware
//
#include <hip/hip_runtime.h>
#include <hip/hip_bf16.h>

// ---- problem constants ----
#define DIM   2048
#define ATTN  512
#define BATCH 256
#define LLEN  196
#define LPAD  256   // 4 groups of 64 rows (rows >=196 zero-padded)

typedef __attribute__((ext_vector_type(16))) __bf16 v16bf;
typedef __attribute__((ext_vector_type(8)))  float  v8f;

__device__ __forceinline__ unsigned short f2bfu(float f) {
    unsigned u = __float_as_uint(f);
    unsigned r = u + 0x7FFFu + ((u >> 16) & 1u);   // round-to-nearest-even
    return (unsigned short)(r >> 16);
}

// ---------------------------------------------------------------------------
// k0: repack W_enc (f32, [d][a] row-major) into bf16 WMMA-B fragments.
// For a k-block of 32, lane t holds N = t%16, K = 16*(t/16) + h (h=0..15).
// Wpack[((atile*64 + kblk)*32 + lane)*16 + h]
// ---------------------------------------------------------------------------
__global__ __launch_bounds__(256) void k0_pack(const float* __restrict__ W_enc,
                                               unsigned short* __restrict__ Wpack) {
    int gid   = blockIdx.x * 256 + threadIdx.x;      // < 1048576
    int h     = gid & 15;
    int lane  = (gid >> 4) & 31;
    int kblk  = (gid >> 9) & 63;
    int atile = gid >> 15;
    int a = atile * 16 + (lane & 15);
    int k = kblk * 32 + ((lane >> 4) << 4) + h;
    Wpack[gid] = f2bfu(W_enc[(size_t)k * ATTN + a]);
}

// ---------------------------------------------------------------------------
// k1: c[n][a] = prev_h[n] . W_dec[:,a] + b_dec[a] + b_enc[a]
// ---------------------------------------------------------------------------
__global__ __launch_bounds__(256) void k1_att2(const float* __restrict__ prev_h,
                                               const float* __restrict__ W_dec,
                                               const float* __restrict__ b_enc,
                                               const float* __restrict__ b_dec,
                                               float* __restrict__ c) {
    int n = blockIdx.x, t = threadIdx.x;
    __shared__ float h[512];
    h[t]       = prev_h[(size_t)n * 512 + t];
    h[t + 256] = prev_h[(size_t)n * 512 + t + 256];
    __syncthreads();
    int a0 = t * 2;
    float acc0 = 0.f, acc1 = 0.f;
    for (int d = 0; d < 512; ++d) {
        float hv = h[d];
        const float* w = W_dec + (size_t)d * ATTN + a0;
        acc0 += hv * w[0];
        acc1 += hv * w[1];
    }
    c[(size_t)n * ATTN + a0]     = acc0 + b_enc[a0]     + b_dec[a0];
    c[(size_t)n * ATTN + a0 + 1] = acc1 + b_enc[a0 + 1] + b_dec[a0 + 1];
}

// ---------------------------------------------------------------------------
// k2: fused  scores[n,l] = relu(feature[n,l,:] @ W_enc + c[n,:]) . W_tot
// Workgroup: 8 waves, one (n, 64-row l-group). Wave w owns a-tiles w*4..w*4+3
// and 4 m-tiles -> v8f acc[4][4] (128 VGPRs). K loop: 64 k-blocks of 32,
// staged 2 at a time through LDS with f32->bf16 conversion into the
// A-fragment swizzle. 32 WMMAs per staged iteration; B fragments stream from
// L2-resident Wpack with an explicit one-deep software pipeline (bf_cur /
// bf_next in distinct registers) so each load's latency is hidden behind the
// WMMA quad consuming the previous fragment.
// ---------------------------------------------------------------------------
__global__ __launch_bounds__(256) void k2_scores(const float* __restrict__ feature,
                                                 const unsigned short* __restrict__ Wpack,
                                                 const float* __restrict__ c,
                                                 const float* __restrict__ Wtot,
                                                 float* __restrict__ scores) {
    const int n    = blockIdx.y;
    const int lg   = blockIdx.x;            // 64-row l group (0..3)
    const int tid  = threadIdx.x;
    const int wave = tid >> 5;
    const int lane = tid & 31;
    const int g    = lane >> 4;             // lane half-group
    const int nn   = lane & 15;             // N (= a within tile) position

    __shared__ __align__(32) unsigned short At[2][4][512]; // [kb][mtile][lane*16+h]
    __shared__ float score_lds[64];

    if (tid < 64) score_lds[tid] = 0.f;

    const unsigned short* wbase =
        Wpack + ((size_t)(wave * 4) * 64 * 32 + lane) * 16;
    // fragment at (at, kblk): wbase + (at*64 + kblk)*32*16 halves
    auto bload = [&](int at, int kblk) -> v16bf {
        return *reinterpret_cast<const v16bf*>(wbase + ((size_t)(at * 64 + kblk) * 32) * 16);
    };

    v8f zero = {};
    v8f acc[4][4];
    #pragma unroll
    for (int mt = 0; mt < 4; ++mt)
        #pragma unroll
        for (int at = 0; at < 4; ++at) acc[mt][at] = zero;

    v16bf bf_cur = bload(0, 0);             // prime the B pipeline

    for (int kp = 0; kp < 32; ++kp) {       // 2 k-blocks (64 K) per iteration
        __syncthreads();
        // ---- stage: 8 tiles (2 kb x 4 mt) = 1024 float4 loads, 4 per thread
        #pragma unroll
        for (int e = 0; e < 4; ++e) {
            int q    = e * 256 + tid;       // 0..1023
            int tile = q >> 7;              // 0..7
            int kb   = tile >> 2, mt = tile & 3;
            int s    = q & 127;
            int dl   = s >> 2;              // destination lane
            int c4   = s & 3;               // which 4-K chunk in that lane
            int gg   = dl >> 4;
            int i0   = c4 * 4;
            int kloc = 16 * gg + 4 * c4;
            int m = dl & 15;
            int l = lg * 64 + mt * 16 + m;
            int kglob = (kp * 2 + kb) * 32 + kloc;
            float4 v = make_float4(0.f, 0.f, 0.f, 0.f);
            if (l < LLEN)
                v = *reinterpret_cast<const float4*>(
                        feature + ((size_t)n * LLEN + l) * DIM + kglob);
            unsigned int p0 = (unsigned int)f2bfu(v.x) | ((unsigned int)f2bfu(v.y) << 16);
            unsigned int p1 = (unsigned int)f2bfu(v.z) | ((unsigned int)f2bfu(v.w) << 16);
            *reinterpret_cast<uint2*>(&At[kb][mt][dl * 16 + i0]) = make_uint2(p0, p1);
        }
        __syncthreads();
        // ---- compute: flat 8 steps (2 kb x 4 at), 4 WMMAs each
        v16bf af[4];
        #pragma unroll
        for (int u = 0; u < 8; ++u) {
            const int kb = u >> 2;
            const int at = u & 3;
            if (at == 0) {
                #pragma unroll
                for (int mt = 0; mt < 4; ++mt)
                    af[mt] = *reinterpret_cast<const v16bf*>(&At[kb][mt][lane * 16]);
            }
            // prefetch next B fragment (crosses into next kp at u==7)
            v16bf bf_next;
            if (u < 7) {
                int nu = u + 1;
                bf_next = bload(nu & 3, kp * 2 + (nu >> 2));
            } else {
                bf_next = bload(0, (kp < 31) ? (kp * 2 + 2) : 0);
            }
            #pragma unroll
            for (int mt = 0; mt < 4; ++mt)
                acc[mt][at] = __builtin_amdgcn_wmma_f32_16x16x32_bf16(
                    false, af[mt], false, bf_cur, (short)0, acc[mt][at], false, false);
            bf_cur = bf_next;
        }
    }

    // ---- epilogue: +bias, relu, dot W_tot, reduce over a
    float p[4][8];
    #pragma unroll
    for (int mt = 0; mt < 4; ++mt)
        #pragma unroll
        for (int j = 0; j < 8; ++j) p[mt][j] = 0.f;

    #pragma unroll
    for (int at = 0; at < 4; ++at) {
        int a = (wave * 4 + at) * 16 + nn;
        float cv = c[(size_t)n * ATTN + a];
        float wt = Wtot[a];
        #pragma unroll
        for (int mt = 0; mt < 4; ++mt)
            #pragma unroll
            for (int j = 0; j < 8; ++j) {
                float v = acc[mt][at][j] + cv;
                v = v > 0.f ? v : 0.f;
                p[mt][j] += v * wt;
            }
    }
    // reduce across the 16 lanes holding the same row (xor bits 0..3)
    #pragma unroll
    for (int off = 1; off < 16; off <<= 1)
        #pragma unroll
        for (int mt = 0; mt < 4; ++mt)
            #pragma unroll
            for (int j = 0; j < 8; ++j)
                p[mt][j] += __shfl_xor(p[mt][j], off, 32);
    if (nn == 0) {
        #pragma unroll
        for (int mt = 0; mt < 4; ++mt)
            #pragma unroll
            for (int j = 0; j < 8; ++j)
                atomicAdd(&score_lds[mt * 16 + 8 * g + j], p[mt][j]);
    }
    __syncthreads();
    if (tid < 64) scores[(size_t)n * LPAD + lg * 64 + tid] = score_lds[tid];
}

// ---------------------------------------------------------------------------
// k3: softmax over L per batch row
// ---------------------------------------------------------------------------
__global__ __launch_bounds__(256) void k3_softmax(const float* __restrict__ scores,
                                                  float* __restrict__ alpha) {
    int n = blockIdx.x, t = threadIdx.x;
    __shared__ float red[256];
    float s = (t < LLEN) ? scores[(size_t)n * LPAD + t] : -3.4e38f;
    red[t] = s;
    __syncthreads();
    for (int off = 128; off > 0; off >>= 1) {
        if (t < off) red[t] = fmaxf(red[t], red[t + off]);
        __syncthreads();
    }
    float mx = red[0];
    __syncthreads();
    float e = (t < LLEN) ? __expf(s - mx) : 0.f;
    red[t] = e;
    __syncthreads();
    for (int off = 128; off > 0; off >>= 1) {
        if (t < off) red[t] += red[t + off];
        __syncthreads();
    }
    float inv = 1.f / red[0];
    if (t < LLEN) alpha[(size_t)n * LLEN + t] = e * inv;
}

// ---------------------------------------------------------------------------
// k4: weighted_feature[n,d] = sum_l alpha[n,l] * feature[n,l,d]
// ---------------------------------------------------------------------------
__global__ __launch_bounds__(256) void k4_weighted(const float* __restrict__ feature,
                                                   const float* __restrict__ alpha,
                                                   float* __restrict__ out) {
    int n  = blockIdx.y;
    int d0 = blockIdx.x * 256 + threadIdx.x;
    __shared__ float al[LLEN];
    if (threadIdx.x < LLEN) al[threadIdx.x] = alpha[(size_t)n * LLEN + threadIdx.x];
    __syncthreads();
    const float* f = feature + (size_t)n * LLEN * DIM + d0;
    float acc = 0.f;
    #pragma unroll 4
    for (int l = 0; l < LLEN; ++l) acc += al[l] * f[(size_t)l * DIM];
    out[(size_t)n * DIM + d0] = acc;
}

// ---------------------------------------------------------------------------
extern "C" void kernel_launch(void* const* d_in, const int* in_sizes, int n_in,
                              void* d_out, int out_size, void* d_ws, size_t ws_size,
                              hipStream_t stream) {
    const float* feature = (const float*)d_in[0];
    const float* prev_h  = (const float*)d_in[1];
    const float* W_enc   = (const float*)d_in[2];
    const float* b_enc   = (const float*)d_in[3];
    const float* W_dec   = (const float*)d_in[4];
    const float* b_dec   = (const float*)d_in[5];
    const float* W_tot   = (const float*)d_in[6];
    // b_tot (d_in[7]) shifts all scores equally -> alpha unchanged; omitted.

    char* ws = (char*)d_ws;
    unsigned short* Wpack = (unsigned short*)(ws);                       // 2 MB
    float* c_bias         = (float*)(ws + 2u * 1024 * 1024);             // 512 KB
    float* scores         = (float*)(ws + 2u * 1024 * 1024 + 512 * 1024);// 256 KB

    float* out_wf    = (float*)d_out;                 // (256, 2048)
    float* out_alpha = out_wf + (size_t)BATCH * DIM;  // (256, 196)

    k0_pack   <<<4096, 256, 0, stream>>>(W_enc, Wpack);
    k1_att2   <<<BATCH, 256, 0, stream>>>(prev_h, W_dec, b_enc, b_dec, c_bias);
    k2_scores <<<dim3(4, BATCH), 256, 0, stream>>>(feature, Wpack, c_bias, W_tot, scores);
    k3_softmax<<<BATCH, 256, 0, stream>>>(scores, out_alpha);
    k4_weighted<<<dim3(DIM / 256, BATCH), 256, 0, stream>>>(feature, out_alpha, out_wf);
}
